// BahdanauAttention_63479616635356
// MI455X (gfx1250) — matrix operand order repaired and checked by reading, not compiled, not run on hardware
//
#include <hip/hip_runtime.h>
#include <hip/hip_bf16.h>
#include <stddef.h>

// ---------- types ----------
typedef __attribute__((ext_vector_type(16))) __bf16     v16bf;
typedef __attribute__((ext_vector_type(8)))  float      v8f;
typedef __attribute__((ext_vector_type(4)))  float      f32x4;
typedef __attribute__((ext_vector_type(4)))  unsigned   u32x4;
typedef __attribute__((ext_vector_type(2)))  unsigned   u32x2;

union BF16x16 { v16bf v; u32x4 u[2]; };

__device__ __forceinline__ unsigned f2bf(float f) {
    unsigned u = __float_as_uint(f);
    return (u + 0x7FFFu + ((u >> 16) & 1u)) >> 16;   // round-to-nearest-even bf16
}

#define BATCH 128
#define SEQ   2048
#define DIM   512

// ---------------------------------------------------------------------------
// 1) Pre-swizzle W_ctx (f32 [512,512] row-major, k-major) into WMMA-B bf16
//    layout: entry[(ntile*16 + kchunk)*32 + lane] = 16 bf16 (32B), where lane
//    element 2j,2j+1 = W[kbase+2j..+1][ntile*16 + (lane&15)],
//    kbase = kchunk*32 + (lane>>4)*16.   (contiguous k order, same as the A loads)
// ---------------------------------------------------------------------------
__global__ __launch_bounds__(256) void bah_wswz(const float* __restrict__ W,
                                                unsigned* __restrict__ Wswz) {
    int u = blockIdx.x * 256 + threadIdx.x;          // 131072 total
    int j    =  u        & 7;
    int lane = (u >> 3)  & 31;
    int c    = (u >> 8)  & 15;
    int t    =  u >> 12;                             // 0..31
    int k = c * 32 + ((lane >> 4) << 4) + 2 * j;
    int n = t * 16 + (lane & 15);
    unsigned lo = f2bf(W[(size_t)k * DIM + n]);
    unsigned hi = f2bf(W[(size_t)(k + 1) * DIM + n]);
    Wswz[((size_t)(t * 16 + c) * 32 + lane) * 8 + j] = lo | (hi << 16);
}

// ---------------------------------------------------------------------------
// 2) qplus[b,n] = input[b]·W_in[:,n] + b_in[n] + b_ctx[n]
// ---------------------------------------------------------------------------
__global__ __launch_bounds__(256) void bah_qplus(const float* __restrict__ input,
                                                 const float* __restrict__ W_in,
                                                 const float* __restrict__ b_in,
                                                 const float* __restrict__ b_ctx,
                                                 float* __restrict__ qplus) {
    int t = blockIdx.x * 256 + threadIdx.x;          // 65536
    int b = t >> 9, n = t & 511;
    float sum = b_in[n] + b_ctx[n];
    const float* in = input + (size_t)b * DIM;
    for (int d = 0; d < DIM; ++d) sum = fmaf(in[d], W_in[(size_t)d * DIM + n], sum);
    qplus[t] = sum;
}

// ---------------------------------------------------------------------------
// 3) scores[b,s] = sum_n w_score[n] * tanh(qplus[b,n] + (context[b,s]·W_ctx)[n])
//    Block = 64 threads (2 waves). Each block: 32 s-rows, bf16 tile in LDS.
//    Each wave: 16-row tile, full N=512, K=512 via v_wmma_f32_16x16x32_bf16.
// ---------------------------------------------------------------------------
#define LROW 520   // padded bf16 row stride (1040B = 260 dwords -> conflict-free b128)

__global__ __launch_bounds__(64) void bah_scores(const float* __restrict__ ctx,
                                                 const float* __restrict__ qplus,
                                                 const float* __restrict__ wscore,
                                                 const unsigned* __restrict__ Wswz,
                                                 float* __restrict__ scores) {
    __shared__ unsigned short lds[32 * LROW];
    const int b   = blockIdx.y;
    const int s0  = blockIdx.x * 32;
    const int tid = threadIdx.x;

    // stage 32x512 f32 context rows -> bf16 LDS (vectorized, coalesced)
    const f32x4* src = reinterpret_cast<const f32x4*>(ctx + ((size_t)b * SEQ + s0) * DIM);
    for (int i = tid; i < 32 * 128; i += 64) {
        int row = i >> 7;
        int q4  = i & 127;
        f32x4 v = src[(size_t)row * 128 + q4];
        u32x2 p;
        p.x = f2bf(v.x) | (f2bf(v.y) << 16);
        p.y = f2bf(v.z) | (f2bf(v.w) << 16);
        *reinterpret_cast<u32x2*>(&lds[row * LROW + (q4 << 2)]) = p;
    }
    __syncthreads();

    const int wv   = tid >> 5;
    const int lane = tid & 31;
    const int m    = lane & 15;     // A row within tile / N within n-tile
    const int kh   = lane >> 4;     // K-half select for A; M-half for C/D
    const unsigned short* arow = lds + (wv * 16 + m) * LROW;

    float part[8] = {0.f, 0.f, 0.f, 0.f, 0.f, 0.f, 0.f, 0.f};
    const v8f vzero = {0.f, 0.f, 0.f, 0.f, 0.f, 0.f, 0.f, 0.f};
    const u32x4* bbase = reinterpret_cast<const u32x4*>(Wswz);

    for (int g = 0; g < 8; ++g) {                     // N groups of 4 tiles (64 cols)
        v8f acc[4] = {vzero, vzero, vzero, vzero};
        for (int c = 0; c < 16; ++c) {                // K chunks of 32
            BF16x16 au;                               // A: 16x32 bf16, contiguous k (same as B)
            const u32x4* ap =
                reinterpret_cast<const u32x4*>(arow + c * 32 + kh * 16);
            au.u[0] = ap[0];                          // K {base..base+7}
            au.u[1] = ap[1];                          // K {base+8..base+15}
#pragma unroll
            for (int t = 0; t < 4; ++t) {
                int ntile = g * 4 + t;
                BF16x16 bu;                           // B: 32x16 bf16, pre-swizzled
                const u32x4* bp = bbase + ((size_t)(ntile * 16 + c) * 32 + lane) * 2;
                bu.u[0] = bp[0];
                bu.u[1] = bp[1];
                acc[t] = __builtin_amdgcn_wmma_f32_16x16x32_bf16(
                    false, au.v, false, bu.v, (short)0, acc[t], false, false);
            }
        }
        // epilogue: w_score[n] * tanh(k + qplus) accumulated per row
#pragma unroll
        for (int t = 0; t < 4; ++t) {
            int N = (g * 4 + t) * 16 + m;
            float qn = qplus[(size_t)b * DIM + N];
            float wn = wscore[N];
#pragma unroll
            for (int r = 0; r < 8; ++r)
                part[r] = fmaf(wn, tanhf(acc[t][r] + qn), part[r]);
        }
    }
    // reduce over the 16 lanes sharing a row set (lanes 0-15: M 0-7; 16-31: M 8-15)
#pragma unroll
    for (int r = 0; r < 8; ++r) {
        float p = part[r];
        p += __shfl_xor(p, 8);
        p += __shfl_xor(p, 4);
        p += __shfl_xor(p, 2);
        p += __shfl_xor(p, 1);
        if (m == 0)
            scores[(size_t)b * SEQ + s0 + wv * 16 + kh * 8 + r] = p;
    }
}

// ---------------------------------------------------------------------------
// 4) masked softmax over S per row; in-place on the attn slice of d_out
// ---------------------------------------------------------------------------
__global__ __launch_bounds__(256) void bah_softmax(float* __restrict__ attn,
                                                   const unsigned char* __restrict__ mask) {
    __shared__ float red[256];
    const int b = blockIdx.x, tid = threadIdx.x;
    float v[8];
    float mx = -__builtin_inff();
#pragma unroll
    for (int i = 0; i < 8; ++i) {
        int s = tid + i * 256;
        float x = attn[(size_t)b * SEQ + s];
        if (mask[(size_t)b * SEQ + s]) x = -__builtin_inff();
        v[i] = x;
        mx = fmaxf(mx, x);
    }
    red[tid] = mx; __syncthreads();
    for (int off = 128; off; off >>= 1) {
        if (tid < off) red[tid] = fmaxf(red[tid], red[tid + off]);
        __syncthreads();
    }
    mx = red[0]; __syncthreads();
    float sum = 0.f;
#pragma unroll
    for (int i = 0; i < 8; ++i) {
        v[i] = __expf(v[i] - mx);                // exp(-inf)=0 handles masked
        sum += v[i];
    }
    red[tid] = sum; __syncthreads();
    for (int off = 128; off; off >>= 1) {
        if (tid < off) red[tid] += red[tid + off];
        __syncthreads();
    }
    float inv = 1.0f / red[0];
#pragma unroll
    for (int i = 0; i < 8; ++i)
        attn[(size_t)b * SEQ + tid + i * 256] = v[i] * inv;
}

// ---------------------------------------------------------------------------
// 5) ctx_vec[b,d] = sum_s attn[b,s]*context[b,s,d]  (s split across 8 blocks)
// ---------------------------------------------------------------------------
__global__ __launch_bounds__(256) void bah_ctxvec(const float* __restrict__ attn,
                                                  const float* __restrict__ ctx,
                                                  float* __restrict__ ctxvec) {
    __shared__ float a[256];
    const int b = blockIdx.y;
    const int s0 = blockIdx.x * 256;
    const int tid = threadIdx.x;
    a[tid] = attn[(size_t)b * SEQ + s0 + tid];
    __syncthreads();
    float acc0 = 0.f, acc1 = 0.f;
    const float* base = ctx + ((size_t)b * SEQ + s0) * DIM;
    for (int s = 0; s < 256; ++s) {
        float w = a[s];
        acc0 = fmaf(w, base[(size_t)s * DIM + tid], acc0);
        acc1 = fmaf(w, base[(size_t)s * DIM + tid + 256], acc1);
    }
    atomicAdd(&ctxvec[(size_t)b * DIM + tid], acc0);
    atomicAdd(&ctxvec[(size_t)b * DIM + tid + 256], acc1);
}

// ---------------------------------------------------------------------------
// 6) x = tanh(concat(ctx_vec, input) @ W_out + b_out)
// ---------------------------------------------------------------------------
__global__ __launch_bounds__(256) void bah_out(const float* __restrict__ ctxvec,
                                               const float* __restrict__ input,
                                               const float* __restrict__ W_out,
                                               const float* __restrict__ b_out,
                                               float* __restrict__ x) {
    int t = blockIdx.x * 256 + threadIdx.x;          // 65536
    int b = t >> 9, n = t & 511;
    float sum = b_out[n];
    const float* cv = ctxvec + (size_t)b * DIM;
    const float* in = input + (size_t)b * DIM;
    for (int d = 0; d < DIM; ++d)
        sum = fmaf(cv[d], W_out[(size_t)d * DIM + n], sum);
    for (int d = 0; d < DIM; ++d)
        sum = fmaf(in[d], W_out[(size_t)(DIM + d) * DIM + n], sum);
    x[t] = tanhf(sum);
}

// ---------------------------------------------------------------------------
extern "C" void kernel_launch(void* const* d_in, const int* in_sizes, int n_in,
                              void* d_out, int out_size, void* d_ws, size_t ws_size,
                              hipStream_t stream) {
    const float*         input   = (const float*)d_in[0];
    const float*         context = (const float*)d_in[1];
    const unsigned char* mask    = (const unsigned char*)d_in[2];
    const float*         W_in    = (const float*)d_in[3];
    const float*         b_in    = (const float*)d_in[4];
    const float*         W_ctx   = (const float*)d_in[5];
    const float*         b_ctx   = (const float*)d_in[6];
    const float*         w_score = (const float*)d_in[7];
    const float*         W_out   = (const float*)d_in[8];
    const float*         b_out   = (const float*)d_in[9];

    float* x    = (float*)d_out;                     // [128,512]
    float* attn = (float*)d_out + BATCH * DIM;       // [128,2048] (scores in-place)

    float*    wsf    = (float*)d_ws;
    float*    qplus  = wsf;                          // 65536 f32
    float*    ctxvec = wsf + 65536;                  // 65536 f32
    unsigned* Wswz   = (unsigned*)(wsf + 131072);    // 131072 u32 (512 KB bf16)

    bah_wswz  <<<512, 256, 0, stream>>>(W_ctx, Wswz);
    bah_qplus <<<256, 256, 0, stream>>>(input, W_in, b_in, b_ctx, qplus);
    bah_scores<<<dim3(64, 128), 64, 0, stream>>>(context, qplus, w_score, Wswz, attn);
    bah_softmax<<<128, 256, 0, stream>>>(attn, mask);
    hipMemsetAsync(ctxvec, 0, BATCH * DIM * sizeof(float), stream);
    bah_ctxvec<<<dim3(8, 128), 256, 0, stream>>>(attn, context, ctxvec);
    bah_out   <<<256, 256, 0, stream>>>(ctxvec, input, W_out, b_out, x);
}
